// BiLSTM_CRF_16518444220901
// MI455X (gfx1250) — matrix operand order repaired and checked by reading, not compiled, not run on hardware
//
#include <hip/hip_runtime.h>

typedef __attribute__((ext_vector_type(16))) _Float16 v16h;
typedef __attribute__((ext_vector_type(8)))  _Float16 v8h;
typedef __attribute__((ext_vector_type(8)))  float    v8f;

#define BATCH 64
#define TLEN  512
#define EDIM  256
#define HDIM  128
#define KTAGS 32
#define G4H   512   // 4*H
#define BT    (BATCH*TLEN)

// ---------------- device helpers ----------------

__device__ __forceinline__ float sigf(float x) {
    return 1.0f / (1.0f + __expf(-x));
}
__device__ __forceinline__ float tanh_fast(float x) {
    x = fminf(fmaxf(x, -12.0f), 12.0f);
    float e = __expf(2.0f * x);
    return (e - 1.0f) / (e + 1.0f);
}

// A fragment: 16x32 f16 tile, row-major source with leading dim lda.
// lanes 0-15 : row = lane,    halves 0-7 -> K k0..k0+7,  halves 8-15 -> K k0+16..k0+23
// lanes16-31 : row = lane-16, halves 0-7 -> K k0+8..+15, halves 8-15 -> K k0+24..+31
__device__ __forceinline__ v16h load_a_frag(const _Float16* A, int lda,
                                            int row0, int k0, int lane) {
    int m  = lane & 15;
    int kh = lane >> 4;
    const _Float16* p = A + (size_t)(row0 + m) * lda + k0 + kh * 8;
    v8h lo = *(const v8h*)p;
    v8h hi = *(const v8h*)(p + 16);
    v16h a;
#pragma unroll
    for (int i = 0; i < 8; ++i) { a[i] = lo[i]; a[i + 8] = hi[i]; }
    return a;
}

// B fragment: 32x16 f16 tile. B[k][n] = W[n0+n][k0+k], W row-major (Ncols x ldw).
// lanes 0-15: N = lane, halves 0-7 -> K k0..k0+7, halves 8-15 -> K k0+16..+23 ; lanes 16-31: K k0+8..+15, k0+24..+31
__device__ __forceinline__ v16h load_b_frag(const _Float16* W, int ldw,
                                            int n0, int k0, int lane) {
    int n  = lane & 15;
    int kh = lane >> 4;
    const _Float16* p = W + (size_t)(n0 + n) * ldw + k0 + kh * 8;
    v8h lo = *(const v8h*)p;
    v8h hi = *(const v8h*)(p + 16);
    v16h b;
#pragma unroll
    for (int i = 0; i < 8; ++i) { b[i] = lo[i]; b[i + 8] = hi[i]; }
    return b;
}

// ---------------- kernels ----------------

__global__ void f32_to_f16_kernel(const float* __restrict__ src,
                                  _Float16* __restrict__ dst, int n) {
    int i = blockIdx.x * blockDim.x + threadIdx.x;
    if (i < n) dst[i] = (_Float16)src[i];
}

// one block per (b,t) row, 256 threads over E
__global__ void embed_kernel(const int* __restrict__ x,
                             const float* __restrict__ emb,
                             _Float16* __restrict__ e16) {
    int row = blockIdx.x;
    int d   = threadIdx.x;
    int tok = x[row];
    e16[(size_t)row * EDIM + d] = (_Float16)emb[(size_t)tok * EDIM + d];
}

// C = A(M x K) * W^T + bias.  W is (Ncols x Kdim) row-major f16.
// Each wave computes TWO adjacent 16x16 ntiles sharing one A fragment.
// mode 0: out[R * (NT*16) + col]   (emissions, R = b*T+t row-major)
// mode 1: out[((R%T)*B + R/T)*512 + col]   (xs scatter to [t][b][4H])
__global__ __launch_bounds__(256)
void gemm_wmma_kernel(const _Float16* __restrict__ A, int lda,
                      const _Float16* __restrict__ W,
                      const float* __restrict__ bias,
                      float* __restrict__ out,
                      int NT, int Kdim, int mode) {
    int lane   = threadIdx.x & 31;
    int gw     = blockIdx.x * 8 + (threadIdx.x >> 5);
    int npairs = NT >> 1;
    int mtile  = gw / npairs;
    int np     = gw % npairs;

    v8f acc0 = {};
    v8f acc1 = {};
    for (int k0 = 0; k0 < Kdim; k0 += 32) {
        v16h a  = load_a_frag(A, lda, mtile * 16, k0, lane);
        v16h b0 = load_b_frag(W, Kdim, (np * 2 + 0) * 16, k0, lane);
        acc0 = __builtin_amdgcn_wmma_f32_16x16x32_f16(
            false, a, false, b0, (short)0, acc0, false, false);
        v16h b1 = load_b_frag(W, Kdim, (np * 2 + 1) * 16, k0, lane);
        acc1 = __builtin_amdgcn_wmma_f32_16x16x32_f16(
            false, a, false, b1, (short)0, acc1, false, false);
    }
    int n  = lane & 15;
    int kh = lane >> 4;
#pragma unroll
    for (int q = 0; q < 2; ++q) {
        int col  = (np * 2 + q) * 16 + n;
        float bv = bias[col];
        const v8f& acc = q ? acc1 : acc0;
#pragma unroll
        for (int r = 0; r < 8; ++r) {
            int row = mtile * 16 + r + 8 * kh;
            size_t idx;
            if (mode == 0) {
                idx = (size_t)row * (NT * 16) + col;
            } else {
                int t  = row & (TLEN - 1);
                int b_ = row >> 9;
                idx = ((size_t)t * BATCH + b_) * G4H + col;
            }
            out[idx] = acc[r] + bv;
        }
    }
}

// One bidirectional LSTM layer. grid=2 (dir), block=512 (16 waves).
// xs: [2][T][B][512] f32 (precomputed x-projection + bias).
// Whh_* : (512,128) f32 global, converted to f16 in LDS once.
// hcat:  [B][T][256] f16 output (fwd cols 0..127, bwd 128..255).
// xs is loaded into registers consumed only after the WMMA K-loop, so the
// per-step 128KB HBM stream overlaps the LDS-fed matrix work.
__global__ __launch_bounds__(512)
void lstm_layer_kernel(const float* __restrict__ xs,
                       const float* __restrict__ WhhF,
                       const float* __restrict__ WhhB,
                       _Float16* __restrict__ hcat) {
    extern __shared__ char smem[];
    _Float16* sWhh = (_Float16*)smem;                       // 512*128 f16 = 128KB
    _Float16* sH   = (_Float16*)(smem + G4H * HDIM * 2);    // 64*128 f16 = 16KB

    int tid  = threadIdx.x;
    int lane = tid & 31;
    int w    = tid >> 5;
    int dir  = blockIdx.x;

    const float* Whh = dir ? WhhB : WhhF;
    for (int i = tid; i < G4H * HDIM; i += 512) sWhh[i] = (_Float16)Whh[i];
    for (int i = tid; i < BATCH * HDIM; i += 512) sH[i] = (_Float16)0.0f;
    __syncthreads();

    int mt[2], jt[2];
#pragma unroll
    for (int p = 0; p < 2; ++p) {
        int id = w * 2 + p;     // 0..31 : 4 mtiles x 8 jtiles
        mt[p] = id >> 3;
        jt[p] = id & 7;
    }
    float cst[2][8];
#pragma unroll
    for (int p = 0; p < 2; ++p)
#pragma unroll
        for (int r = 0; r < 8; ++r) cst[p][r] = 0.0f;

    int n  = lane & 15;
    int kh = lane >> 4;
    const float* xbase = xs + (size_t)dir * TLEN * BATCH * G4H;

    for (int s = 0; s < TLEN; ++s) {
        int t = dir ? (TLEN - 1 - s) : s;
        const float* xt = xbase + (size_t)t * BATCH * G4H;

        // xs preactivations -> registers (no consumer until elementwise stage)
        v8f xv[2][4];
#pragma unroll
        for (int p = 0; p < 2; ++p)
#pragma unroll
            for (int g = 0; g < 4; ++g) {
                int colb = (g * 8 + jt[p]) * 16 + n;
#pragma unroll
                for (int r = 0; r < 8; ++r) {
                    int row = mt[p] * 16 + r + 8 * kh;
                    xv[p][g][r] = xt[(size_t)row * G4H + colb];
                }
            }

        v8f acc[2][4];
#pragma unroll
        for (int p = 0; p < 2; ++p)
#pragma unroll
            for (int g = 0; g < 4; ++g) acc[p][g] = (v8f){};

#pragma unroll
        for (int kk = 0; kk < 4; ++kk) {
            v16h a0 = load_a_frag(sH, HDIM, mt[0] * 16, kk * 32, lane);
            v16h a1 = load_a_frag(sH, HDIM, mt[1] * 16, kk * 32, lane);
#pragma unroll
            for (int g = 0; g < 4; ++g) {
                v16h b0 = load_b_frag(sWhh, HDIM, (g * 8 + jt[0]) * 16, kk * 32, lane);
                acc[0][g] = __builtin_amdgcn_wmma_f32_16x16x32_f16(
                    false, a0, false, b0, (short)0, acc[0][g], false, false);
                v16h b1 = load_b_frag(sWhh, HDIM, (g * 8 + jt[1]) * 16, kk * 32, lane);
                acc[1][g] = __builtin_amdgcn_wmma_f32_16x16x32_f16(
                    false, a1, false, b1, (short)0, acc[1][g], false, false);
            }
        }
        __syncthreads();   // everyone done reading sH (h_{s-1})

#pragma unroll
        for (int p = 0; p < 2; ++p) {
            int jcol = jt[p] * 16 + n;
#pragma unroll
            for (int r = 0; r < 8; ++r) {
                int row = mt[p] * 16 + r + 8 * kh;   // batch index
                float iv = acc[p][0][r] + xv[p][0][r];
                float fv = acc[p][1][r] + xv[p][1][r];
                float gv = acc[p][2][r] + xv[p][2][r];
                float ov = acc[p][3][r] + xv[p][3][r];
                float c  = sigf(fv) * cst[p][r] + sigf(iv) * tanh_fast(gv);
                cst[p][r] = c;
                float h  = sigf(ov) * tanh_fast(c);
                _Float16 h16 = (_Float16)h;
                sH[row * HDIM + jcol] = h16;
                hcat[((size_t)row * TLEN + t) * 256 + dir * HDIM + jcol] = h16;
            }
        }
        __syncthreads();   // new h visible before next step's reads
    }
}

// CRF NLL: single block, 1024 threads. emissions [B][T][K] f32.
__global__ __launch_bounds__(1024)
void crf_kernel(const float* __restrict__ emis,
                const int* __restrict__ tags,
                const float* __restrict__ trans,
                const float* __restrict__ start,
                const float* __restrict__ endv,
                float* __restrict__ out) {
    __shared__ float sTT[KTAGS * KTAGS];        // transposed: sTT[kp][k]
    __shared__ float sAl[2][BATCH * KTAGS];
    __shared__ float sPart[BATCH];

    int tid = threadIdx.x;
    {   // trans^T load (1024 elems)
        int kp = tid >> 5, k = tid & 31;
        sTT[kp * KTAGS + k] = trans[k * KTAGS + kp];
    }
#pragma unroll
    for (int q = 0; q < 2; ++q) {
        int e = tid + q * 1024;
        int b = e >> 5, k = e & 31;
        sAl[0][e] = start[k] + emis[(size_t)b * TLEN * KTAGS + k];
    }
    __syncthreads();

    int par = 0;
    for (int t = 1; t < TLEN; ++t) {
#pragma unroll
        for (int q = 0; q < 2; ++q) {
            int e = tid + q * 1024;
            int b = e >> 5, kp = e & 31;
            const float* al = sAl[par] + b * KTAGS;
            const float* tt = sTT + kp * KTAGS;
            float m = -3.0e38f;
#pragma unroll 8
            for (int k = 0; k < KTAGS; ++k) m = fmaxf(m, al[k] + tt[k]);
            float ssum = 0.0f;
#pragma unroll 8
            for (int k = 0; k < KTAGS; ++k) ssum += __expf(al[k] + tt[k] - m);
            sAl[par ^ 1][e] = m + __logf(ssum) +
                              emis[((size_t)b * TLEN + t) * KTAGS + kp];
        }
        __syncthreads();
        par ^= 1;
    }

    if (tid < BATCH) {
        const float* al = sAl[par] + tid * KTAGS;
        float m = -3.0e38f;
        for (int k = 0; k < KTAGS; ++k) m = fmaxf(m, al[k] + endv[k]);
        float ssum = 0.0f;
        for (int k = 0; k < KTAGS; ++k) ssum += __expf(al[k] + endv[k] - m);
        float logZ = m + __logf(ssum);

        const int* tg = tags + (size_t)tid * TLEN;
        float gold = start[tg[0]] + emis[(size_t)tid * TLEN * KTAGS + tg[0]];
        for (int t = 1; t < TLEN; ++t) {
            gold += emis[((size_t)tid * TLEN + t) * KTAGS + tg[t]] +
                    trans[tg[t] * KTAGS + tg[t - 1]];
        }
        gold += endv[tg[TLEN - 1]];
        sPart[tid] = logZ - gold;
    }
    __syncthreads();
    if (tid == 0) {
        float s = 0.0f;
        for (int b = 0; b < BATCH; ++b) s += sPart[b];
        out[0] = s / (float)BATCH;
    }
}

// ---------------- host ----------------

extern "C" void kernel_launch(void* const* d_in, const int* in_sizes, int n_in,
                              void* d_out, int out_size, void* d_ws, size_t ws_size,
                              hipStream_t stream) {
    (void)in_sizes; (void)n_in; (void)out_size; (void)ws_size;

    const int*   x      = (const int*)d_in[0];
    const int*   tags   = (const int*)d_in[2];
    const float* emb    = (const float*)d_in[4];
    const float* Wih[4] = { (const float*)d_in[5],  (const float*)d_in[8],
                            (const float*)d_in[11], (const float*)d_in[14] };
    const float* Whh[4] = { (const float*)d_in[6],  (const float*)d_in[9],
                            (const float*)d_in[12], (const float*)d_in[15] };
    const float* bia[4] = { (const float*)d_in[7],  (const float*)d_in[10],
                            (const float*)d_in[13], (const float*)d_in[16] };
    const float* fcW    = (const float*)d_in[17];
    const float* fcb    = (const float*)d_in[18];
    const float* trans  = (const float*)d_in[19];
    const float* start  = (const float*)d_in[20];
    const float* endv   = (const float*)d_in[21];
    float* out = (float*)d_out;

    char* ws = (char*)d_ws;
    size_t off = 0;
    auto alloc = [&](size_t bytes) -> char* {
        char* p = ws + off;
        off += (bytes + 255) & ~(size_t)255;
        return p;
    };
    _Float16* e16    = (_Float16*)alloc((size_t)BT * EDIM * 2);        // 16 MB
    _Float16* hcat0  = (_Float16*)alloc((size_t)BT * 256 * 2);         // 16 MB
    _Float16* hcat1  = (_Float16*)alloc((size_t)BT * 256 * 2);         // 16 MB
    float*    xsbuf  = (float*)alloc((size_t)2 * TLEN * BATCH * G4H * 4); // 128 MB
    _Float16* wih16[4];
    for (int i = 0; i < 4; ++i) wih16[i] = (_Float16*)alloc((size_t)G4H * EDIM * 2);
    _Float16* fcW16  = (_Float16*)alloc((size_t)KTAGS * 256 * 2);
    float*    emis   = (float*)alloc((size_t)BT * KTAGS * 4);          // 4 MB

    // weight conversions to f16
    for (int i = 0; i < 4; ++i) {
        int nel = G4H * EDIM;
        f32_to_f16_kernel<<<(nel + 255) / 256, 256, 0, stream>>>(Wih[i], wih16[i], nel);
    }
    {
        int nel = KTAGS * 256;
        f32_to_f16_kernel<<<(nel + 255) / 256, 256, 0, stream>>>(fcW, fcW16, nel);
    }

    // embedding gather -> f16
    embed_kernel<<<BT, EDIM, 0, stream>>>(x, emb, e16);

    const size_t xs_dir = (size_t)TLEN * BATCH * G4H;
    // two ntiles per wave -> tiles/2 waves, 8 waves per block
    const int proj_blocks = (BT / 16) * ((G4H / 16) / 2) / 8;   // 4096
    const size_t lstm_shmem = (size_t)G4H * HDIM * 2 + (size_t)BATCH * HDIM * 2;

    // ---- layer 0 ----
    for (int dir = 0; dir < 2; ++dir)
        gemm_wmma_kernel<<<proj_blocks, 256, 0, stream>>>(
            e16, EDIM, wih16[dir], bia[dir], xsbuf + dir * xs_dir,
            G4H / 16, EDIM, /*mode=*/1);
    lstm_layer_kernel<<<2, 512, lstm_shmem, stream>>>(xsbuf, Whh[0], Whh[1], hcat0);

    // ---- layer 1 ----
    for (int dir = 0; dir < 2; ++dir)
        gemm_wmma_kernel<<<proj_blocks, 256, 0, stream>>>(
            hcat0, 256, wih16[2 + dir], bia[2 + dir], xsbuf + dir * xs_dir,
            G4H / 16, 256, /*mode=*/1);
    lstm_layer_kernel<<<2, 512, lstm_shmem, stream>>>(xsbuf, Whh[2], Whh[3], hcat1);

    // ---- FC -> emissions ----
    {
        int blocks = (BT / 16) * ((KTAGS / 16) / 2) / 8;   // 256
        gemm_wmma_kernel<<<blocks, 256, 0, stream>>>(
            hcat1, 256, fcW16, fcb, emis, KTAGS / 16, 256, /*mode=*/0);
    }

    // ---- CRF NLL ----
    crf_kernel<<<1, 1024, 0, stream>>>(emis, tags, trans, start, endv, out);
}
